// LocalAttention_86320252715346
// MI455X (gfx1250) — matrix operand order repaired and checked by reading, not compiled, not run on hardware
//
#include <hip/hip_runtime.h>

#define HIDDEN   1024
#define NUM_HEADS 16
#define HEAD_DIM  64
#define WINDOW    64
#define BATCH      2
#define SEQ     4096
#define NW (SEQ / WINDOW)
#define KSTEP 32

typedef __bf16 bf16_t;
typedef __attribute__((ext_vector_type(16))) __bf16 v16bf;
typedef __attribute__((ext_vector_type(8)))  float  v8f;
typedef __attribute__((ext_vector_type(4)))  int    i32x4;

#ifndef __has_builtin
#define __has_builtin(x) 0
#endif
#if __has_builtin(__builtin_amdgcn_global_load_async_to_lds_b128)
#define HAVE_ASYNC_LDS 1
#else
#define HAVE_ASYNC_LDS 0
#endif

// ---- 16B memory->LDS copy: async DMA (ASYNCcnt) when available, else via VGPRs ----
// Probe-discovered signature: (int4 AS1* gsrc, int4 AS3* ldst, imm offset, imm cpol)
__device__ inline void cp_b128(const bf16_t* g, bf16_t* l) {
#if HAVE_ASYNC_LDS
  __builtin_amdgcn_global_load_async_to_lds_b128(
      (__attribute__((address_space(1))) i32x4*)g,
      (__attribute__((address_space(3))) i32x4*)l, 0, 0);
#else
  *(uint4*)l = *(const uint4*)g;
#endif
}

__device__ inline void cp_join() {
#if HAVE_ASYNC_LDS
#if __has_builtin(__builtin_amdgcn_s_wait_asynccnt)
  __builtin_amdgcn_s_wait_asynccnt(0);
#else
  asm volatile("s_wait_asynccnt 0x0" ::: "memory");
#endif
#endif
}

// ---------------- WMMA 16x16x32 bf16 fragment helpers (contiguous k = 16*halfK + s order) ----------------

// A-matrix 16x32 (MxK), 16-bit elements.
// lane = 16*halfK + m ; halfK=0 -> K 0..15, halfK=1 -> K 16..31
__device__ inline v16bf frag_a_bf16(const bf16_t* src, int ld, int lane) {
  const int m  = lane & 15;
  const int kb = (lane >> 4) << 4;
  const bf16_t* p = src + m * ld + kb;
  v16bf f;
#pragma unroll
  for (int j = 0; j < 8; ++j) f[j] = p[j];           // K = kb .. kb+7
#pragma unroll
  for (int j = 0; j < 8; ++j) f[8 + j] = p[8 + j];   // K = kb+8 .. kb+15
  return f;
}

// Same A layout, but source is f32 (converted on load).
__device__ inline v16bf frag_a_f32(const float* src, int ld, int lane) {
  const int m  = lane & 15;
  const int kb = (lane >> 4) << 4;
  const float* p = src + m * ld + kb;
  v16bf f;
#pragma unroll
  for (int j = 0; j < 8; ++j) f[j] = (bf16_t)p[j];
#pragma unroll
  for (int j = 0; j < 8; ++j) f[8 + j] = (bf16_t)p[8 + j];
  return f;
}

// B-matrix 32x16 (KxN) with memory rows of N, K contiguous: src[n*ld + k]
// (weights W[N,K] row-major => Bmat[k][n] = W[n][k]).
// lane = 16*halfK + n ; halfK=0 -> K 0..15, halfK=1 -> K 16..31
__device__ inline v16bf frag_b_rowNK(const bf16_t* src, int ld, int lane) {
  const int n  = lane & 15;
  const int kb = (lane >> 4) << 4;
  const bf16_t* p = src + n * ld + kb;
  v16bf f;
#pragma unroll
  for (int i = 0; i < 16; ++i) f[i] = p[i];
  return f;
}

// B-matrix 32x16 (KxN) where memory is row-major [K][N]: src[k*ld + n] (V tile).
__device__ inline v16bf frag_b_rowKN(const bf16_t* src, int ld, int lane) {
  const int n  = lane & 15;
  const int kb = (lane >> 4) << 4;
  const bf16_t* p = src + kb * ld + n;
  v16bf f;
#pragma unroll
  for (int i = 0; i < 16; ++i) f[i] = p[i * ld];
  return f;
}

// ---------------- elementwise kernels ----------------

__global__ void f32_to_bf16_kernel(const float* __restrict__ src, bf16_t* __restrict__ dst,
                                   size_t n) {
  size_t i  = (size_t)blockIdx.x * blockDim.x + threadIdx.x;
  size_t st = (size_t)gridDim.x * blockDim.x;
  for (; i < n; i += st) dst[i] = (bf16_t)src[i];
}

__global__ void zero_f4_kernel(float4* __restrict__ p, size_t n4) {
  size_t i  = (size_t)blockIdx.x * blockDim.x + threadIdx.x;
  size_t st = (size_t)gridDim.x * blockDim.x;
  const float4 z = make_float4(0.f, 0.f, 0.f, 0.f);
  for (; i < n4; i += st) p[i] = z;
}

// ---------------- GEMM:  Y[M,N] = A[M,K] @ W[N,K]^T + bias[N] ----------------
// 128x128 block tile, 256 threads (8 waves). Wave -> 2 row-tiles x 4 col-tiles.
// Double-buffered LDS staging through async memory->LDS DMA.
__global__ __launch_bounds__(256) void gemm_bias_kernel(
    const bf16_t* __restrict__ A, const bf16_t* __restrict__ W,
    const float* __restrict__ bias,
    float* __restrict__ Cf, bf16_t* __restrict__ Cb,   // exactly one non-null
    int M, int N, int K)
{
  __shared__ __align__(16) bf16_t As[2][128 * KSTEP];
  __shared__ __align__(16) bf16_t Bs[2][128 * KSTEP];

  const int m0   = blockIdx.y * 128;
  const int n0   = blockIdx.x * 128;
  const int tid  = threadIdx.x;
  const int wave = tid >> 5;
  const int lane = tid & 31;
  const int wr   = (wave & 3) * 2;   // row tiles wr, wr+1   (of 8)
  const int wc   = (wave >> 2) * 4;  // col tiles wc..wc+3   (of 8)

  v8f acc[2][4] = {};

  // Each thread moves 2 x b128 per matrix per stage (512 x 16B per 128x32 tile).
  const int e0 = tid, e1 = tid + 256;
  const int r0 = e0 >> 2, s0 = (e0 & 3) << 3;
  const int r1 = e1 >> 2, s1 = (e1 & 3) << 3;

  auto stage = [&](int kk, int buf) {
    cp_b128(A + (size_t)(m0 + r0) * K + kk + s0, &As[buf][r0 * KSTEP + s0]);
    cp_b128(A + (size_t)(m0 + r1) * K + kk + s1, &As[buf][r1 * KSTEP + s1]);
    cp_b128(W + (size_t)(n0 + r0) * K + kk + s0, &Bs[buf][r0 * KSTEP + s0]);
    cp_b128(W + (size_t)(n0 + r1) * K + kk + s1, &Bs[buf][r1 * KSTEP + s1]);
  };

  stage(0, 0);
  cp_join();
  __syncthreads();

  int buf = 0;
  for (int kk = 0; kk < K; kk += KSTEP) {
    if (kk + KSTEP < K) stage(kk + KSTEP, buf ^ 1);  // prefetch next stage

    v16bf a0 = frag_a_bf16(&As[buf][(wr + 0) * 16 * KSTEP], KSTEP, lane);
    v16bf a1 = frag_a_bf16(&As[buf][(wr + 1) * 16 * KSTEP], KSTEP, lane);
#pragma unroll
    for (int c = 0; c < 4; ++c) {
      v16bf b = frag_b_rowNK(&Bs[buf][(wc + c) * 16 * KSTEP], KSTEP, lane);
      acc[0][c] = __builtin_amdgcn_wmma_f32_16x16x32_bf16(false, a0, false, b, (short)0,
                                                          acc[0][c], false, false);
      acc[1][c] = __builtin_amdgcn_wmma_f32_16x16x32_bf16(false, a1, false, b, (short)0,
                                                          acc[1][c], false, false);
    }
    cp_join();
    __syncthreads();
    buf ^= 1;
  }

  const int rbase = (lane >> 4) << 3;
  const int nl    = lane & 15;
#pragma unroll
  for (int rt = 0; rt < 2; ++rt) {
    const int mloc = (wr + rt) * 16 + rbase;
#pragma unroll
    for (int c = 0; c < 4; ++c) {
      const int n    = n0 + (wc + c) * 16 + nl;
      const float bv = bias[n];
#pragma unroll
      for (int r = 0; r < 8; ++r) {
        const size_t m = (size_t)(m0 + mloc + r);
        const float x  = acc[rt][c][r] + bv;
        if (Cf) Cf[m * N + n] = x;
        else    Cb[m * N + n] = (bf16_t)x;
      }
    }
  }
}

// ---------------- Windowed attention: one block per (batch, window) ----------------
__global__ __launch_bounds__(256) void attn_kernel(
    const bf16_t* __restrict__ Qp, const bf16_t* __restrict__ Kp,
    const bf16_t* __restrict__ Vp,
    bf16_t* __restrict__ Att,        // [B*S, HIDDEN] attended (bf16, for Wo GEMM)
    float* __restrict__ AttnMap)     // [B, S, S]
{
  __shared__ __align__(16) bf16_t Qs[64 * 64];
  __shared__ __align__(16) bf16_t Ks[64 * 64];
  __shared__ __align__(16) bf16_t Vs[64 * 64];
  __shared__ __align__(16) float  P [64 * 64];
  __shared__ __align__(16) float  Pm[64 * 64];

  const int bi  = blockIdx.x / NW;
  const int wnd = blockIdx.x % NW;
  const int w0  = wnd * WINDOW;
  const int tid = threadIdx.x;
  const int wave = tid >> 5;
  const int lane = tid & 31;
  const int t0   = wave * 2;
  const int tm   = t0 >> 2;
  const int tn0  = t0 & 3;
  const int tn1  = tn0 + 1;
  const float scale = 0.125f;  // 1/sqrt(64)

  for (int h = 0; h < NUM_HEADS; ++h) {
    // Stage 64x64 Q/K/V head tiles into LDS: 512 x b128 per tile, 2 per thread per tile.
    {
#pragma unroll
      for (int t = 0; t < 2; ++t) {
        const int e   = tid + t * 256;
        const int row = e >> 3;
        const int sg  = (e & 7) << 3;   // 8 bf16 per b128
        const size_t gb = (size_t)(bi * SEQ + w0 + row) * HIDDEN + h * HEAD_DIM + sg;
        cp_b128(Qp + gb, Qs + row * 64 + sg);
        cp_b128(Kp + gb, Ks + row * 64 + sg);
        cp_b128(Vp + gb, Vs + row * 64 + sg);
      }
      cp_join();
    }
    __syncthreads();

    // scores = (Q @ K^T) * scale   -> P (f32, LDS)
    v8f sA = {};
    v8f sB = {};
#pragma unroll
    for (int kc = 0; kc < HEAD_DIM; kc += 32) {
      v16bf a  = frag_a_bf16(Qs + tm * 16 * 64 + kc, 64, lane);
      v16bf b0 = frag_b_rowNK(Ks + tn0 * 16 * 64 + kc, 64, lane);
      v16bf b1 = frag_b_rowNK(Ks + tn1 * 16 * 64 + kc, 64, lane);
      sA = __builtin_amdgcn_wmma_f32_16x16x32_bf16(false, a, false, b0, (short)0, sA, false, false);
      sB = __builtin_amdgcn_wmma_f32_16x16x32_bf16(false, a, false, b1, (short)0, sB, false, false);
    }
    {
      const int mloc = tm * 16 + ((lane >> 4) << 3);
      const int nl = lane & 15;
#pragma unroll
      for (int r = 0; r < 8; ++r) {
        P[(mloc + r) * 64 + tn0 * 16 + nl] = sA[r] * scale;
        P[(mloc + r) * 64 + tn1 * 16 + nl] = sB[r] * scale;
      }
    }
    __syncthreads();

    // softmax over each of the 64 rows (threads 0..63, one row each)
    if (tid < 64) {
      float* row = P + tid * 64;
      float mx = row[0];
#pragma unroll 8
      for (int i = 1; i < 64; ++i) mx = fmaxf(mx, row[i]);
      float sum = 0.f;
#pragma unroll 8
      for (int i = 0; i < 64; ++i) { float e = __expf(row[i] - mx); row[i] = e; sum += e; }
      const float inv = 1.0f / sum;
#pragma unroll 8
      for (int i = 0; i < 64; ++i) row[i] *= inv;
    }
    __syncthreads();

    // deterministic head-mean accumulation
    {
      const float c = 1.0f / (float)NUM_HEADS;
      for (int idx = tid; idx < 64 * 64; idx += 256)
        Pm[idx] = (h == 0) ? P[idx] * c : Pm[idx] + P[idx] * c;
    }

    // attended = P @ V  (A from f32 P with on-the-fly bf16 convert)
    v8f oA = {};
    v8f oB = {};
#pragma unroll
    for (int kc = 0; kc < WINDOW; kc += 32) {
      v16bf a  = frag_a_f32(P + tm * 16 * 64 + kc, 64, lane);
      v16bf b0 = frag_b_rowKN(Vs + kc * 64 + tn0 * 16, 64, lane);
      v16bf b1 = frag_b_rowKN(Vs + kc * 64 + tn1 * 16, 64, lane);
      oA = __builtin_amdgcn_wmma_f32_16x16x32_bf16(false, a, false, b0, (short)0, oA, false, false);
      oB = __builtin_amdgcn_wmma_f32_16x16x32_bf16(false, a, false, b1, (short)0, oB, false, false);
    }
    {
      const int mloc = tm * 16 + ((lane >> 4) << 3);
      const int nl = lane & 15;
#pragma unroll
      for (int r = 0; r < 8; ++r) {
        const size_t ro = (size_t)(bi * SEQ + w0 + mloc + r) * HIDDEN + h * HEAD_DIM;
        Att[ro + tn0 * 16 + nl] = (bf16_t)oA[r];
        Att[ro + tn1 * 16 + nl] = (bf16_t)oB[r];
      }
    }
    __syncthreads();  // protect LDS reuse next head; also publishes Pm for the epilogue
  }

  // write mean-probability block onto the attention-map diagonal
  for (int idx = tid; idx < 64 * 64; idx += 256) {
    const int qr = idx >> 6, kcol = idx & 63;
    AttnMap[(size_t)bi * SEQ * SEQ + (size_t)(w0 + qr) * SEQ + (w0 + kcol)] = Pm[idx];
  }
}

// ---------------- host launcher ----------------

extern "C" void kernel_launch(void* const* d_in, const int* in_sizes, int n_in,
                              void* d_out, int out_size, void* d_ws, size_t ws_size,
                              hipStream_t stream)
{
  (void)in_sizes; (void)n_in; (void)out_size; (void)ws_size;

  const float* q  = (const float*)d_in[0];
  const float* k  = (const float*)d_in[1];
  const float* v  = (const float*)d_in[2];
  const float* Wq = (const float*)d_in[3];
  const float* bq = (const float*)d_in[4];
  const float* Wk = (const float*)d_in[5];
  const float* bk = (const float*)d_in[6];
  const float* Wv = (const float*)d_in[7];
  const float* bv = (const float*)d_in[8];
  const float* Wo = (const float*)d_in[9];
  const float* bo = (const float*)d_in[10];

  float* out  = (float*)d_out;                               // [B,S,HIDDEN]
  float* attn = out + (size_t)BATCH * SEQ * HIDDEN;          // [B,S,S]

  const size_t MSZ = (size_t)BATCH * SEQ * HIDDEN;           // 8,388,608
  const size_t WSZ = (size_t)HIDDEN * HIDDEN;                // 1,048,576

  char* w = (char*)d_ws;
  bf16_t* q_bf  = (bf16_t*)w; w += MSZ * sizeof(bf16_t);
  bf16_t* k_bf  = (bf16_t*)w; w += MSZ * sizeof(bf16_t);
  bf16_t* v_bf  = (bf16_t*)w; w += MSZ * sizeof(bf16_t);
  bf16_t* Wq_bf = (bf16_t*)w; w += WSZ * sizeof(bf16_t);
  bf16_t* Wk_bf = (bf16_t*)w; w += WSZ * sizeof(bf16_t);
  bf16_t* Wv_bf = (bf16_t*)w; w += WSZ * sizeof(bf16_t);
  bf16_t* Wo_bf = (bf16_t*)w; w += WSZ * sizeof(bf16_t);
  bf16_t* qp    = (bf16_t*)w; w += MSZ * sizeof(bf16_t);
  bf16_t* kp    = (bf16_t*)w; w += MSZ * sizeof(bf16_t);
  bf16_t* vp    = (bf16_t*)w; w += MSZ * sizeof(bf16_t);
  bf16_t* att   = (bf16_t*)w; w += MSZ * sizeof(bf16_t);

  // 1) convert activations + weights to bf16
  f32_to_bf16_kernel<<<4096, 256, 0, stream>>>(q,  q_bf,  MSZ);
  f32_to_bf16_kernel<<<4096, 256, 0, stream>>>(k,  k_bf,  MSZ);
  f32_to_bf16_kernel<<<4096, 256, 0, stream>>>(v,  v_bf,  MSZ);
  f32_to_bf16_kernel<<<1024, 256, 0, stream>>>(Wq, Wq_bf, WSZ);
  f32_to_bf16_kernel<<<1024, 256, 0, stream>>>(Wk, Wk_bf, WSZ);
  f32_to_bf16_kernel<<<1024, 256, 0, stream>>>(Wv, Wv_bf, WSZ);
  f32_to_bf16_kernel<<<1024, 256, 0, stream>>>(Wo, Wo_bf, WSZ);

  // 2) projections: x @ W^T + b  (emit bf16)
  const int M = BATCH * SEQ;
  dim3 gGemm(HIDDEN / 128, M / 128);
  gemm_bias_kernel<<<gGemm, 256, 0, stream>>>(q_bf, Wq_bf, bq, nullptr, qp, M, HIDDEN, HIDDEN);
  gemm_bias_kernel<<<gGemm, 256, 0, stream>>>(k_bf, Wk_bf, bk, nullptr, kp, M, HIDDEN, HIDDEN);
  gemm_bias_kernel<<<gGemm, 256, 0, stream>>>(v_bf, Wv_bf, bv, nullptr, vp, M, HIDDEN, HIDDEN);

  // 3) zero the attention-map region (block-diagonal blocks overwritten next)
  zero_f4_kernel<<<8192, 256, 0, stream>>>((float4*)attn, (size_t)BATCH * SEQ * SEQ / 4);

  // 4) windowed attention + head-mean map
  attn_kernel<<<BATCH * NW, 256, 0, stream>>>(qp, kp, vp, att, attn);

  // 5) output projection into d_out (f32)
  gemm_bias_kernel<<<gGemm, 256, 0, stream>>>(att, Wo_bf, bo, out, nullptr, M, HIDDEN, HIDDEN);
}
